// LSTMCell_71957882077215
// MI455X (gfx1250) — compile-verified, matrix operand order checked by reading, not run on hardware
//
#include <hip/hip_runtime.h>

typedef __attribute__((ext_vector_type(16))) __bf16 v16bf;
typedef __attribute__((ext_vector_type(8)))  __bf16 bf16x8;
typedef __attribute__((ext_vector_type(4)))  __bf16 bf16x4;
typedef __attribute__((ext_vector_type(8)))  float  v8f;
typedef int v4i_vs __attribute__((vector_size(16)));   // matches builtin param type

#define BATCH 4096
#define HID   1024
#define KTOT  2048
#define BM    128
#define BN    64           // columns per gate per workgroup
#define BK    64           // K-slab per LDS stage (2 WMMA K-steps)
#define NKT   (KTOT / BK)  // 32
#define LDSS  72           // padded K-stride in bf16 elems (144B = 9*16B, 36 bank-words)

#if defined(__has_builtin)
#if __has_builtin(__builtin_amdgcn_global_load_async_to_lds_b128)
#define USE_ASYNC_COPY 1
#endif
#endif

__device__ __forceinline__ void cp16(void* lds, const void* gsrc) {
#ifdef USE_ASYNC_COPY
    __builtin_amdgcn_global_load_async_to_lds_b128(
        (__attribute__((address_space(1))) v4i_vs*)gsrc,
        (__attribute__((address_space(3))) v4i_vs*)lds, 0, 0);
#else
    *(uint4*)lds = *(const uint4*)gsrc;
#endif
}

__device__ __forceinline__ void wait_async_all() {
#ifdef USE_ASYNC_COPY
#if __has_builtin(__builtin_amdgcn_s_wait_asynccnt)
    __builtin_amdgcn_s_wait_asynccnt(0);
#else
    asm volatile("s_wait_asynccnt 0" ::: "memory");
#endif
#endif
}

__device__ __forceinline__ float sigmoid_f(float x) {
    return 1.0f / (1.0f + __expf(-x));
}
__device__ __forceinline__ float tanh_f(float x) {
    float e = __expf(2.0f * x);
    return (e - 1.0f) / (e + 1.0f);
}

// ---------------- pack kernel 1: A = bf16([x | h_prev])  [4096][2048] ----------------
__global__ __launch_bounds__(256)
void pack_a(const float* __restrict__ x, const float* __restrict__ h,
            __bf16* __restrict__ A)
{
    int f4  = blockIdx.x * 256 + threadIdx.x;   // float4 units over 4096*2048
    int row = f4 >> 9;                          // 512 float4 per 2048-wide row
    int k   = (f4 & 511) * 4;
    const float* src = (k < 1024) ? (x + (size_t)row * 1024 + k)
                                  : (h + (size_t)row * 1024 + (k - 1024));
    float4 v = *(const float4*)src;
    bf16x4 b;
    b[0] = (__bf16)v.x; b[1] = (__bf16)v.y; b[2] = (__bf16)v.z; b[3] = (__bf16)v.w;
    *(bf16x4*)(A + (size_t)row * KTOT + k) = b;
}

// ------- pack kernel 2: Bt[g][n][k] = bf16([W_g ; U_g]^T), k-contiguous (transpose) -------
__global__ __launch_bounds__(256)
void pack_w(const float* __restrict__ Wf, const float* __restrict__ Wi,
            const float* __restrict__ Wc, const float* __restrict__ Wo,
            const float* __restrict__ Uf, const float* __restrict__ Ui,
            const float* __restrict__ Uc, const float* __restrict__ Uo,
            __bf16* __restrict__ Bt)
{
    __shared__ __bf16 t[64 * LDSS];
    const int tid = threadIdx.x;
    const int g  = blockIdx.z;
    const int k0 = blockIdx.y * 64;     // 0..2048
    const int n0 = blockIdx.x * 64;     // 0..1024
    const float* src;
    int krel;
    if (k0 < 1024) {
        krel = k0;
        src = (g == 0) ? Wf : (g == 1) ? Wi : (g == 2) ? Wc : Wo;
    } else {
        krel = k0 - 1024;
        src = (g == 0) ? Uf : (g == 1) ? Ui : (g == 2) ? Uc : Uo;
    }
    #pragma unroll
    for (int j = 0; j < 4; ++j) {                      // load 64k x 64n fp32 tile
        int f4  = j * 256 + tid;
        int kk  = f4 >> 4;
        int nn4 = f4 & 15;
        float4 v = *(const float4*)(src + (size_t)(krel + kk) * HID + n0 + nn4 * 4);
        bf16x4 b;
        b[0] = (__bf16)v.x; b[1] = (__bf16)v.y; b[2] = (__bf16)v.z; b[3] = (__bf16)v.w;
        *(bf16x4*)&t[kk * LDSS + nn4 * 4] = b;
    }
    __syncthreads();
    #pragma unroll
    for (int j = 0; j < 4; ++j) {                      // store transposed, k-contiguous
        int f4 = j * 256 + tid;
        int nn = f4 >> 4;
        int kq = f4 & 15;
        bf16x4 b;
        #pragma unroll
        for (int i = 0; i < 4; ++i) b[i] = t[(kq * 4 + i) * LDSS + nn];
        *(bf16x4*)(Bt + ((size_t)g * HID + n0 + nn) * KTOT + k0 + kq * 4) = b;
    }
}

// ---------------- fused LSTM GEMM: gates = A @ [W;U] per 4 gates + epilogue ----------------
__global__ __launch_bounds__(256, 1)
void lstm_gemm(const __bf16* __restrict__ A,     // [4096][2048]
               const __bf16* __restrict__ Bt,    // [4][1024 n][2048 k]
               const float* __restrict__ cprev,
               const float* __restrict__ bfb, const float* __restrict__ bib,
               const float* __restrict__ bcb, const float* __restrict__ bob,
               float* __restrict__ hout, float* __restrict__ cout)
{
    __shared__ __bf16 sA[2][BM * LDSS];        // [buf][row][k]
    __shared__ __bf16 sB[2][4 * BN * LDSS];    // [buf][gate*64+n][k]

    const int tid     = threadIdx.x;
    const int lane    = tid & 31;
    const int waveId  = tid >> 5;
    const int waveRow = waveId & 3;            // 4 row-waves x 32 rows
    const int waveCol = waveId >> 2;           // 2 col-waves x 32 cols
    const int half    = lane >> 4;
    const int m       = lane & 15;
    const int kb      = half * 8;              // per-lane K chunk base (ISA 16-bit A/B layout)

    const int m0 = blockIdx.y * BM;
    const int n0 = blockIdx.x * BN;

    v8f acc[4][2][2];
    #pragma unroll
    for (int g = 0; g < 4; ++g)
        #pragma unroll
        for (int rt = 0; rt < 2; ++rt)
            #pragma unroll
            for (int ct = 0; ct < 2; ++ct)
                #pragma unroll
                for (int e = 0; e < 8; ++e)
                    acc[g][rt][ct][e] = 0.0f;

    auto stage = [&](int kt, int buf) {
        const int k0 = kt * BK;
        #pragma unroll
        for (int j = 0; j < 4; ++j) {          // A: 128 rows x 64 k = 1024 16B chunks
            int c = j * 256 + tid;
            int row = c >> 3;
            int kc  = c & 7;
            cp16(&sA[buf][row * LDSS + kc * 8],
                 A + (size_t)(m0 + row) * KTOT + k0 + kc * 8);
        }
        #pragma unroll
        for (int g = 0; g < 4; ++g)            // B: 4 x (64 n x 64 k) = 2048 16B chunks
            #pragma unroll
            for (int j = 0; j < 2; ++j) {
                int c = j * 256 + tid;
                int n  = c >> 3;
                int kc = c & 7;
                cp16(&sB[buf][(g * BN + n) * LDSS + kc * 8],
                     Bt + ((size_t)g * HID + n0 + n) * KTOT + k0 + kc * 8);
            }
    };

    stage(0, 0);

    for (int kt = 0; kt < NKT; ++kt) {
        const int buf = kt & 1;
        wait_async_all();          // my stage(kt) copies have landed
        __syncthreads();           // everyone's stage(kt) copies visible
        if (kt + 1 < NKT)
            stage(kt + 1, buf ^ 1);  // overlaps with WMMA below

        #pragma unroll
        for (int s = 0; s < 2; ++s) {          // two 16x16x32 K-steps per slab
            v16bf afrag[2];
            #pragma unroll
            for (int rt = 0; rt < 2; ++rt) {
                const __bf16* pa =
                    &sA[buf][(waveRow * 32 + rt * 16 + m) * LDSS + s * 32 + kb];
                bf16x8 lo = *(const bf16x8*)pa;
                bf16x8 hi = *(const bf16x8*)(pa + 16);
                afrag[rt] = __builtin_shufflevector(lo, hi,
                    0,1,2,3,4,5,6,7,8,9,10,11,12,13,14,15);
            }
            #pragma unroll
            for (int g = 0; g < 4; ++g) {
                #pragma unroll
                for (int ct = 0; ct < 2; ++ct) {
                    const __bf16* pb =
                        &sB[buf][(g * BN + waveCol * 32 + ct * 16 + m) * LDSS + s * 32 + kb];
                    bf16x8 lo = *(const bf16x8*)pb;
                    bf16x8 hi = *(const bf16x8*)(pb + 16);
                    v16bf bfrag = __builtin_shufflevector(lo, hi,
                        0,1,2,3,4,5,6,7,8,9,10,11,12,13,14,15);
                    #pragma unroll
                    for (int rt = 0; rt < 2; ++rt) {
                        acc[g][rt][ct] = __builtin_amdgcn_wmma_f32_16x16x32_bf16(
                            false, afrag[rt], false, bfrag,
                            (short)0, acc[g][rt][ct], false, false);
                    }
                }
            }
        }
    }

    // ---- fused LSTM epilogue: bias + activations + state update, in-register ----
    const float* bg[4] = {bfb, bib, bcb, bob};
    #pragma unroll
    for (int ct = 0; ct < 2; ++ct) {
        const int col = n0 + waveCol * 32 + ct * 16 + m;
        const float bF = bg[0][col];
        const float bI = bg[1][col];
        const float bC = bg[2][col];
        const float bO = bg[3][col];
        #pragma unroll
        for (int rt = 0; rt < 2; ++rt) {
            #pragma unroll
            for (int v = 0; v < 8; ++v) {
                const int    row = m0 + waveRow * 32 + rt * 16 + half * 8 + v;
                const size_t idx = (size_t)row * HID + col;
                const float fg = sigmoid_f(acc[0][rt][ct][v] + bF);
                const float ig = sigmoid_f(acc[1][rt][ct][v] + bI);
                const float gg = tanh_f   (acc[2][rt][ct][v] + bC);
                const float og = sigmoid_f(acc[3][rt][ct][v] + bO);
                const float c  = fg * cprev[idx] + ig * gg;
                hout[idx] = og * tanh_f(c);
                cout[idx] = c;
            }
        }
    }
}

extern "C" void kernel_launch(void* const* d_in, const int* in_sizes, int n_in,
                              void* d_out, int out_size, void* d_ws, size_t ws_size,
                              hipStream_t stream) {
    (void)in_sizes; (void)n_in; (void)out_size; (void)ws_size;

    const float* x     = (const float*)d_in[0];
    const float* hprev = (const float*)d_in[1];
    const float* cprev = (const float*)d_in[2];
    const float* Wf = (const float*)d_in[3];
    const float* Uf = (const float*)d_in[4];
    const float* bf = (const float*)d_in[5];
    const float* Wi = (const float*)d_in[6];
    const float* Ui = (const float*)d_in[7];
    const float* bi = (const float*)d_in[8];
    const float* Wc = (const float*)d_in[9];
    const float* Uc = (const float*)d_in[10];
    const float* bc = (const float*)d_in[11];
    const float* Wo = (const float*)d_in[12];
    const float* Uo = (const float*)d_in[13];
    const float* bo = (const float*)d_in[14];

    // workspace layout: A bf16 [4096][2048] (16 MiB) | Bt bf16 [4][1024][2048] (16 MiB)
    __bf16* Abf = (__bf16*)d_ws;
    __bf16* Btw = Abf + (size_t)BATCH * KTOT;

    pack_a<<<(BATCH * KTOT / 4) / 256, 256, 0, stream>>>(x, hprev, Abf);
    pack_w<<<dim3(HID / 64, KTOT / 64, 4), 256, 0, stream>>>(
        Wf, Wi, Wc, Wo, Uf, Ui, Uc, Uo, Btw);

    float* hout = (float*)d_out;                 // tuple (h, c) concatenated flat
    float* cout = hout + (size_t)BATCH * HID;

    lstm_gemm<<<dim3(HID / BN, BATCH / BM), 256, 0, stream>>>(
        Abf, Btw, cprev, bf, bi, bc, bo, hout, cout);
}
